// MultiheadLatentAttentionTensorCast_5411658793258
// MI455X (gfx1250) — matrix operand order repaired and checked by reading, not compiled, not run on hardware
//
#include <hip/hip_runtime.h>
#include <hip/hip_bf16.h>
#include <math.h>
#include <stdint.h>

// MLA (DeepSeek-style) forward for gfx1250.
// GEMMs + attention via v_wmma_f32_16x16x32_bf16; attention tiles staged with
// the Tensor Data Mover (tensor_load_to_lds, TENSORcnt); GEMM staging uses
// global_prefetch for the next K tile.
// B=1, S=2048, D=4096, H=32, Q_LORA=1536, KV_LORA=512, NOPE=128, ROPE=64, V_HEAD=128.

#define EPS_RMS 1e-6f

typedef __attribute__((ext_vector_type(16))) __bf16 bf16x16;
typedef __attribute__((ext_vector_type(8)))  float floatx8;
typedef __attribute__((ext_vector_type(4)))  unsigned int u32x4;
typedef __attribute__((ext_vector_type(8)))  int i32x8;
typedef __attribute__((ext_vector_type(4)))  int i32x4;

__device__ __forceinline__ floatx8 wmma_bf16f32(bf16x16 a, bf16x16 b, floatx8 c) {
  return __builtin_amdgcn_wmma_f32_16x16x32_bf16(false, a, false, b, (short)0, c,
                                                 false, false);
}

// A-fragment (16x32, 16-bit): lane half hi=0 holds K=0..15 contiguous,
// hi=1 holds K=16..31 (matches ld_fragB).
__device__ __forceinline__ bf16x16 ld_fragA(const __bf16* p, int hi) {
  bf16x16 v;
  const __bf16* a = p + 16 * hi;
#pragma unroll
  for (int i = 0; i < 16; ++i) v[i] = a[i];
  return v;
}

// B-fragment (32x16, 16-bit): lanes 0-15 hold K=0..15 contiguous, lanes 16-31
// hold K=16..31 (this code's order, same as ld_fragA).
__device__ __forceinline__ bf16x16 ld_fragB(const __bf16* p, int hi) {
  bf16x16 v;
  const __bf16* a = p + 16 * hi;
#pragma unroll
  for (int i = 0; i < 16; ++i) v[i] = a[i];
  return v;
}

// ---------------------------------------------------------------------------
// Tensor Data Mover: async 2D tile load global -> LDS (D# per ISA 08 §8.3/8.4).
// tile_w elems per row, tile_h rows, row stride in elements, data_size code
// (0=1B,1=2B,2=4B,3=8B). Issued once per calling wave; tracked by TENSORcnt.
// ---------------------------------------------------------------------------
__device__ __forceinline__ void tdm_load_2d(unsigned lds_off, const void* gptr,
                                            unsigned tile_w, unsigned tile_h,
                                            unsigned long long stride_elems,
                                            unsigned data_size) {
  unsigned long long ga = (unsigned long long)(uintptr_t)gptr;
  u32x4 g0;
  g0[0] = 1u;                                   // count=1, user descriptor
  g0[1] = lds_off;                              // LDS byte address
  g0[2] = (unsigned)(ga & 0xffffffffu);         // global_addr[31:0]
  g0[3] = (unsigned)((ga >> 32) & 0x01ffffffu)  // global_addr[56:32]
          | 0x80000000u;                        // type=2 ("image")

  i32x8 g1 = {};
  g1[0] = (int)(data_size << 16);               // wg_mask=0, data_size
  g1[1] = (int)((tile_w & 0xffffu) << 16);      // tensor_dim0[15:0]   @ bits 63:48
  g1[2] = (int)((tile_w >> 16) |                // tensor_dim0[31:16]  @ bits 79:64
                ((tile_h & 0xffffu) << 16));    // tensor_dim1[15:0]   @ bits 111:96? (95:80)
  g1[3] = (int)((tile_h >> 16) |                // tensor_dim1[31:16]
                (tile_w << 16));                // tile_dim0 @ bits 127:112
  g1[4] = (int)(tile_h & 0xffffu);              // tile_dim1 @ bits 143:128, tile_dim2=0
  g1[5] = (int)(unsigned)(stride_elems & 0xffffffffu);        // dim0_stride[31:0]
  g1[6] = (int)(unsigned)((stride_elems >> 32) & 0xffffu);    // dim0_stride[47:32]
  g1[7] = 0;

  i32x4 z4 = {};
#if defined(__clang_major__) && (__clang_major__ >= 23)
  i32x8 z8 = {};
  __builtin_amdgcn_tensor_load_to_lds(g0, g1, z4, z4, z8, 0);
#else
  __builtin_amdgcn_tensor_load_to_lds(g0, g1, z4, z4, 0);
#endif
}

__device__ __forceinline__ unsigned lds_off_of(const void* p) {
  return (unsigned)(uintptr_t)p;  // low 32 bits of flat shared addr = LDS offset
}

// ---------------------------------------------------------------------------
// Generic tiled WMMA GEMM:  C[m,n] = out_scale * sum_k A[m,k] * B(n,k)
//   B_KN = false : B is (N,K) row-major (X @ W^T)
//   B_KN = true  : B is (K,N) row-major (X @ W), transposed while staging
// 64 x BN block tile (BN in {64,128}), BK=32, 256 threads (8 waves), batched
// over blockIdx.z. Requires M%64==0, N%BN==0, K%32==0 (true for all calls).
// ---------------------------------------------------------------------------
template <int BN, bool B_KN, bool OUT_BF16>
__global__ __launch_bounds__(256)
void gemm_wmma_kernel(const float* __restrict__ A, long lda, long aBatch,
                      const float* __restrict__ B, long ldb, long bBatch,
                      void* __restrict__ Cv, long ldc, long cBatch,
                      int K, float out_scale) {
  constexpr int SUB = BN / 32;  // 16-wide subtiles per wave
  __shared__ alignas(32) __bf16 As[64 * 32];
  __shared__ alignas(32) __bf16 Bs[BN * 32];

  const int tid  = threadIdx.x;
  const int lane = tid & 31;
  const int wave = tid >> 5;
  const int hi   = (lane >> 4) & 1;
  const int ln   = lane & 15;

  const long bm = (long)blockIdx.y * 64;
  const long bn = (long)blockIdx.x * BN;
  A += (long)blockIdx.z * aBatch;
  B += (long)blockIdx.z * bBatch;

  const int rb = wave >> 1;              // 16-row block 0..3
  const int cw = (wave & 1) * (BN / 2);  // column base for this wave

  floatx8 acc[SUB] = {};

  for (int k0 = 0; k0 < K; k0 += 32) {
    __syncthreads();
    {  // stage A tile [64 rows][32 k], fp32 -> bf16
      int r = tid >> 2, ks = (tid & 3) * 8;
      const float* src = A + (bm + r) * lda + k0 + ks;
      if (k0 + 32 < K) __builtin_prefetch(src + 32, 0, 1);
      __bf16* dst = &As[r * 32 + ks];
#pragma unroll
      for (int i = 0; i < 8; ++i) dst[i] = (__bf16)src[i];
    }
    if (!B_KN) {  // B (N,K): contiguous K reads
      constexpr int TPR = 256 / BN;  // threads per B row
      constexpr int E   = 32 / TPR;  // elems per thread
      int n = tid / TPR, ks = (tid % TPR) * E;
      const float* src = B + (bn + n) * ldb + k0 + ks;
      if (k0 + 32 < K) __builtin_prefetch(src + 32, 0, 1);
      __bf16* dst = &Bs[n * 32 + ks];
#pragma unroll
      for (int i = 0; i < E; ++i) dst[i] = (__bf16)src[i];
    } else {      // B (K,N): transpose while staging
      constexpr int E = BN / 8;
      int k = tid >> 3, ns = (tid & 7) * E;
      const float* src = B + (long)(k0 + k) * ldb + bn + ns;
      if (k0 + 32 < K) __builtin_prefetch(src + 32 * ldb, 0, 1);
#pragma unroll
      for (int i = 0; i < E; ++i) Bs[(ns + i) * 32 + k] = (__bf16)src[i];
    }
    __syncthreads();

    bf16x16 a = ld_fragA(&As[(rb * 16 + ln) * 32], hi);
#pragma unroll
    for (int s2 = 0; s2 < SUB; ++s2) {
      bf16x16 b = ld_fragB(&Bs[(cw + s2 * 16 + ln) * 32], hi);
      acc[s2] = wmma_bf16f32(a, b, acc[s2]);
    }
  }

  const long cbase = (long)blockIdx.z * cBatch;
#pragma unroll
  for (int s2 = 0; s2 < SUB; ++s2) {
#pragma unroll
    for (int j = 0; j < 8; ++j) {
      long m   = bm + rb * 16 + j + 8 * hi;
      long col = bn + cw + s2 * 16 + ln;
      float v  = acc[s2][j] * out_scale;
      if (OUT_BF16) ((__bf16*)Cv)[cbase + m * ldc + col] = (__bf16)v;
      else          ((float*)Cv)[cbase + m * ldc + col] = v;
    }
  }
}

// ---------------------------------------------------------------------------
// In-place row RMSNorm: x = x * rsqrt(mean(x^2)+eps) * g, one block per row.
// ---------------------------------------------------------------------------
__global__ __launch_bounds__(256)
void rmsnorm_kernel(float* __restrict__ X, const float* __restrict__ g, int L) {
  __shared__ float red[256];
  float* x = X + (long)blockIdx.x * L;
  float ss = 0.f;
  for (int i = threadIdx.x; i < L; i += 256) { float v = x[i]; ss += v * v; }
  red[threadIdx.x] = ss;
  __syncthreads();
  for (int s = 128; s > 0; s >>= 1) {
    if (threadIdx.x < s) red[threadIdx.x] += red[threadIdx.x + s];
    __syncthreads();
  }
  float inv = rsqrtf(red[0] / (float)L + EPS_RMS);
  for (int i = threadIdx.x; i < L; i += 256) x[i] = x[i] * inv * g[i];
}

// ---------------------------------------------------------------------------
// kv post: kcat[s, 0:512]  = bf16(rmsnorm(kv[s,0:512], g_kv_a))
//          kcat[s, 512:576]= bf16(rope(kv[s,512:576], cos[s], sin[s]))
// ---------------------------------------------------------------------------
__global__ __launch_bounds__(256)
void kv_post_kernel(const float* __restrict__ kv, const float* __restrict__ g,
                    const float* __restrict__ cosT, const float* __restrict__ sinT,
                    __bf16* __restrict__ kcat) {
  __shared__ float red[256];
  const long s = blockIdx.x;
  const float* row = kv + s * 576;
  float ss = 0.f;
  for (int i = threadIdx.x; i < 512; i += 256) { float v = row[i]; ss += v * v; }
  red[threadIdx.x] = ss;
  __syncthreads();
  for (int st = 128; st > 0; st >>= 1) {
    if (threadIdx.x < st) red[threadIdx.x] += red[threadIdx.x + st];
    __syncthreads();
  }
  float inv = rsqrtf(red[0] / 512.0f + EPS_RMS);
  __bf16* out = kcat + s * 576;
  for (int i = threadIdx.x; i < 512; i += 256) out[i] = (__bf16)(row[i] * inv * g[i]);
  if (threadIdx.x < 64) {
    int i = threadIdx.x;
    float x   = row[512 + i];
    float rot = (i < 32) ? -row[512 + i + 32] : row[512 + i - 32];
    out[512 + i] = (__bf16)(x * cosT[s * 64 + i] + rot * sinT[s * 64 + i]);
  }
}

// ---------------------------------------------------------------------------
// q rope: qcat[s,h,512+i] = bf16(rope(q[s, h*192+128 + i]) * scale)
// ---------------------------------------------------------------------------
__global__ __launch_bounds__(64)
void q_rope_kernel(const float* __restrict__ q, const float* __restrict__ cosT,
                   const float* __restrict__ sinT, __bf16* __restrict__ qcat,
                   float scale) {
  const long s = blockIdx.x;
  const long h = blockIdx.y;
  const int  i = threadIdx.x;
  const float* base = q + s * 6144 + h * 192 + 128;
  float x   = base[i];
  float rot = (i < 32) ? -base[i + 32] : base[i - 32];
  float v   = (x * cosT[s * 64 + i] + rot * sinT[s * 64 + i]) * scale;
  qcat[(s * 32 + h) * 576 + 512 + i] = (__bf16)v;
}

// ---------------------------------------------------------------------------
// Flash-style MLA attention core (softmax scale pre-folded into qcat):
//   S[q,k] = qcat[q,h,:576] . kcat[k,:576]   (causal)
//   ctx_latent[q,h,:512] = softmax(S) @ kcat[:, :512]
// One workgroup = (16 q-rows, one head), 4 waves (128 thr). Q and K tiles are
// staged row-major into LDS by the Tensor Data Mover (wave 0 issues
// tensor_load_to_lds, waits on TENSORcnt, block barrier publishes the tile).
// ---------------------------------------------------------------------------
__global__ __launch_bounds__(128)
void mla_attention_kernel(const __bf16* __restrict__ qcat,
                          const __bf16* __restrict__ kcat,
                          float* __restrict__ ctx_latent) {
  __shared__ alignas(32) __bf16 Qs[16 * 576];  // [row][dim]
  __shared__ alignas(32) __bf16 Ks[32 * 576];  // [key][dim]
  __shared__ alignas(32) float  Ps[16 * 32];   // score tile (fp32)
  __shared__ alignas(32) __bf16 Pb[16 * 32];   // prob tile (bf16)
  __shared__ float mrow[16], lrow[16], crow[16];

  const int tid  = threadIdx.x;
  const int lane = tid & 31;
  const int wave = tid >> 5;
  const int hi   = (lane >> 4) & 1;
  const int ln   = lane & 15;
  const int qb   = blockIdx.x;  // q block of 16 rows
  const int h    = blockIdx.y;  // head

  // TDM: Q tile = 16 rows x 576 bf16, consecutive q rows are 32*576 elems apart.
  if (wave == 0)
    tdm_load_2d(lds_off_of(Qs), qcat + ((long)(qb * 16) * 32 + h) * 576,
                576, 16, 18432ull, 1);
  if (tid < 16) { mrow[tid] = -3.0e38f; lrow[tid] = 0.f; }

  floatx8 acc[8] = {};  // ctx accumulators: cols [128*wave + 16*s2, +16)

  const int nkt = (qb >> 1) + 1;  // causal: 32-key tiles up to the diagonal
  for (int kt = 0; kt < nkt; ++kt) {
    __syncthreads();
    if (wave == 0) {
      // TDM: K tile = 32 keys x 576 bf16, contiguous rows.
      tdm_load_2d(lds_off_of(Ks), kcat + (long)kt * 32 * 576, 576, 32, 576ull, 1);
      __builtin_amdgcn_s_wait_tensorcnt(0);  // covers Q tile too (in-order)
    }
    Ps[tid] = 0.f; Ps[tid + 128] = 0.f; Ps[tid + 256] = 0.f; Ps[tid + 384] = 0.f;
    __syncthreads();

    // Score pass: all 4 waves; wave = (key-col half jn) x (576-dim half kh).
    {
      const int jn = wave & 1;
      const int kh = wave >> 1;
      floatx8 sa = {};
      for (int step = 0; step < 9; ++step) {
        int k0 = 288 * kh + step * 32;
        bf16x16 a = ld_fragA(&Qs[ln * 576 + k0], hi);
        bf16x16 b = ld_fragB(&Ks[(16 * jn + ln) * 576 + k0], hi);  // contiguous
        sa = wmma_bf16f32(a, b, sa);
      }
#pragma unroll
      for (int j = 0; j < 8; ++j)
        atomicAdd(&Ps[(j + 8 * hi) * 32 + 16 * jn + ln], sa[j]);
    }
    __syncthreads();

    // Online-softmax update: one thread per q row.
    if (tid < 16) {
      const int qidx = qb * 16 + tid;
      float mi = mrow[tid];
      float mnew = mi;
      for (int c = 0; c < 32; ++c) {
        int k = kt * 32 + c;
        float sv = (k <= qidx) ? Ps[tid * 32 + c] : -3.0e38f;
        mnew = fmaxf(mnew, sv);
      }
      float corr = __expf(mi - mnew);
      float sum = 0.f;
      for (int c = 0; c < 32; ++c) {
        int k = kt * 32 + c;
        float p = (k <= qidx) ? __expf(Ps[tid * 32 + c] - mnew) : 0.f;
        Pb[tid * 32 + c] = (__bf16)p;
        sum += p;
      }
      mrow[tid] = mnew;
      lrow[tid] = lrow[tid] * corr + sum;
      crow[tid] = corr;
    }
    __syncthreads();

    // P @ V: all 4 waves, wave covers c-cols [128*wave, +128).
    float cm[8];
#pragma unroll
    for (int j = 0; j < 8; ++j) cm[j] = crow[j + 8 * hi];
#pragma unroll
    for (int s2 = 0; s2 < 8; ++s2)
#pragma unroll
      for (int j = 0; j < 8; ++j) acc[s2][j] *= cm[j];

    bf16x16 pf = ld_fragA(&Pb[ln * 32], hi);
#pragma unroll
    for (int s2 = 0; s2 < 8; ++s2) {
      int c0 = 128 * wave + 16 * s2;
      bf16x16 vf;  // V column c0+ln over 32 keys (strided gather from Ks)
#pragma unroll
      for (int i = 0; i < 16; ++i) vf[i] = Ks[(16 * hi + i) * 576 + c0 + ln];
      acc[s2] = wmma_bf16f32(pf, vf, acc[s2]);
    }
  }

  __syncthreads();
  float lm[8];
#pragma unroll
  for (int j = 0; j < 8; ++j) lm[j] = 1.0f / lrow[j + 8 * hi];
#pragma unroll
  for (int s2 = 0; s2 < 8; ++s2) {
    int c0 = 128 * wave + 16 * s2;
#pragma unroll
    for (int j = 0; j < 8; ++j) {
      long m = (long)qb * 16 + j + 8 * hi;
      ctx_latent[(m * 32 + h) * 512 + c0 + ln] = acc[s2][j] * lm[j];
    }
  }
}

// ---------------------------------------------------------------------------
// Host launcher
// ---------------------------------------------------------------------------
extern "C" void kernel_launch(void* const* d_in, const int* in_sizes, int n_in,
                              void* d_out, int out_size, void* d_ws, size_t ws_size,
                              hipStream_t stream) {
  (void)in_sizes; (void)n_in; (void)out_size; (void)ws_size;

  const float* hidden = (const float*)d_in[0];
  const float* cosT   = (const float*)d_in[1];
  const float* sinT   = (const float*)d_in[2];
  const float* w_q_a  = (const float*)d_in[3];
  const float* g_q_a  = (const float*)d_in[4];
  const float* w_q_b  = (const float*)d_in[5];
  const float* w_kv_a = (const float*)d_in[6];
  const float* g_kv_a = (const float*)d_in[7];
  const float* w_uk_t = (const float*)d_in[8];
  const float* w_uv   = (const float*)d_in[9];
  const float* w_o    = (const float*)d_in[10];

  char* ws = (char*)d_ws;
  size_t off = 0;
  auto take = [&](size_t bytes) {
    char* p = ws + off;
    off = (off + bytes + 255) & ~(size_t)255;
    return p;
  };
  float*  q_a  = (float*) take((size_t)2048 * 1536 * 4);      // q_a (pre+post norm)
  float*  qbuf = (float*) take((size_t)2048 * 6144 * 4);      // q = q_a @ w_q_b^T
  float*  kv   = (float*) take((size_t)2048 * 576 * 4);       // kv_a output
  float*  ctx  = (float*) take((size_t)2048 * 4096 * 4);      // per-head ctx merged
  float*  ctxl = (float*) take((size_t)2048 * 32 * 512 * 4);  // ctx_latent
  __bf16* kcat = (__bf16*)take((size_t)2048 * 576 * 2);       // [kv_c | k_rot] bf16
  __bf16* qcat = (__bf16*)take((size_t)2048 * 32 * 576 * 2);  // [q_latent | q_rope]*scale

  const float scale = 0.07216878364870323f;  // 1/sqrt(192)
  dim3 blk(256);

  // 1) q_a = hidden @ w_q_a^T          [2048,1536] K=4096
  gemm_wmma_kernel<128, false, false><<<dim3(1536 / 128, 32, 1), blk, 0, stream>>>(
      hidden, 4096, 0, w_q_a, 4096, 0, q_a, 1536, 0, 4096, 1.0f);
  // 2) rmsnorm(q_a) in place
  rmsnorm_kernel<<<2048, blk, 0, stream>>>(q_a, g_q_a, 1536);
  // 3) q = q_a @ w_q_b^T               [2048,6144] K=1536
  gemm_wmma_kernel<128, false, false><<<dim3(6144 / 128, 32, 1), blk, 0, stream>>>(
      q_a, 1536, 0, w_q_b, 1536, 0, qbuf, 6144, 0, 1536, 1.0f);
  // 4) kv = hidden @ w_kv_a^T          [2048,576] K=4096  (N%128!=0 -> BN=64)
  gemm_wmma_kernel<64, false, false><<<dim3(576 / 64, 32, 1), blk, 0, stream>>>(
      hidden, 4096, 0, w_kv_a, 4096, 0, kv, 576, 0, 4096, 1.0f);
  // 5) kcat = [rmsnorm(kv_c) | rope(k_rot)]  (bf16)
  kv_post_kernel<<<2048, blk, 0, stream>>>(kv, g_kv_a, cosT, sinT, kcat);
  // 6) qcat[...,512:576] = rope(q_rope) * scale  (bf16)
  q_rope_kernel<<<dim3(2048, 32), dim3(64), 0, stream>>>(qbuf, cosT, sinT, qcat, scale);
  // 7) qcat[...,0:512] = (q_nope @ w_uk_t[h]) * scale, per head (B is (K=128,N=512))
  gemm_wmma_kernel<128, true, true><<<dim3(512 / 128, 32, 32), blk, 0, stream>>>(
      qbuf, 6144, 192, w_uk_t, 512, 128 * 512, qcat, 18432, 576, 128, scale);
  // 8) attention -> ctx_latent [s,h,512]
  mla_attention_kernel<<<dim3(128, 32), dim3(128), 0, stream>>>(qcat, kcat, ctxl);
  // 9) ctx[:, h*128:(h+1)*128] = ctx_latent[:,h,:] @ w_uv[h]  (B is (K=512,N=128))
  gemm_wmma_kernel<128, true, false><<<dim3(128 / 128, 32, 32), blk, 0, stream>>>(
      ctxl, 16384, 512, w_uv, 128, 512 * 128, ctx, 4096, 128, 512, 1.0f);
  // 10) out = ctx @ w_o^T              [2048,4096] K=4096
  gemm_wmma_kernel<128, false, false><<<dim3(4096 / 128, 32, 1), blk, 0, stream>>>(
      ctx, 4096, 0, w_o, 4096, 0, d_out, 4096, 0, 4096, 1.0f);
}
